// CrossPhaseModule_89850715832853
// MI455X (gfx1250) — matrix operand order repaired and checked by reading, not compiled, not run on hardware
//
#include <hip/hip_runtime.h>

#define NN 6
#define BB 4
#define TT 1024
#define DD 768
#define PPH 3
#define HH 2
#define DH 32
#define II 64
#define SS (PPH*TT)   /* 3072 */
#define LN_EPS 1e-5f

typedef __attribute__((ext_vector_type(16))) __bf16 v16bf;
typedef __attribute__((ext_vector_type(8)))  float  v8f;

union V16 {
  v16bf v;
  uint4 q[2];
};

// bf16 elements [0..7] and [16..23] -> one WMMA operand
static __device__ __forceinline__ v16bf ldbf16x16(const __bf16* p) {
  V16 r;
  r.q[0] = *(const uint4*)(p);
  r.q[1] = *(const uint4*)(p + 16);
  return r.v;
}

// A-operand slices: this lane needs elements [0..7] and [16..23] of its K-row
static __device__ __forceinline__ v16bf a_from_bf16(const __bf16* p) {
  V16 r;
  r.q[0] = *(const uint4*)(p);
  r.q[1] = *(const uint4*)(p + 16);
  return r.v;
}

static __device__ __forceinline__ v16bf a_from_f32(const float* p) {
  float4 f0 = *(const float4*)(p);
  float4 f1 = *(const float4*)(p + 4);
  float4 f2 = *(const float4*)(p + 16);
  float4 f3 = *(const float4*)(p + 20);
  v16bf r;
  r[0]  = (__bf16)f0.x; r[1]  = (__bf16)f0.y; r[2]  = (__bf16)f0.z; r[3]  = (__bf16)f0.w;
  r[4]  = (__bf16)f1.x; r[5]  = (__bf16)f1.y; r[6]  = (__bf16)f1.z; r[7]  = (__bf16)f1.w;
  r[8]  = (__bf16)f2.x; r[9]  = (__bf16)f2.y; r[10] = (__bf16)f2.z; r[11] = (__bf16)f2.w;
  r[12] = (__bf16)f3.x; r[13] = (__bf16)f3.y; r[14] = (__bf16)f3.z; r[15] = (__bf16)f3.w;
  return r;
}

static __device__ __forceinline__ v8f wmma_bf16(v16bf a, v16bf b, v8f c) {
  return __builtin_amdgcn_wmma_f32_16x16x32_bf16(false, a, false, b, (short)0, c, false, false);
}

static __device__ __forceinline__ v8f v8zero() {
  v8f z = {0.f,0.f,0.f,0.f,0.f,0.f,0.f,0.f};
  return z;
}

// ---------------- Kernel 1: weight transpose + bf16 convert ----------------
__global__ void prep_weights(const float* __restrict__ Wq, const float* __restrict__ Wk,
                             const float* __restrict__ Wv, const float* __restrict__ Wo,
                             __bf16* __restrict__ Wq_t, __bf16* __restrict__ Wk_t,
                             __bf16* __restrict__ Wv_t, __bf16* __restrict__ Wo_t) {
  int t = blockIdx.x * blockDim.x + threadIdx.x;
  const int TOT = NN * DD * II;
  if (t < TOT) {
    int n = t / (DD * II);
    int rem = t - n * (DD * II);
    int d = rem / II;
    int i = rem - d * II;
    Wq_t[(n * II + i) * DD + d] = (__bf16)Wq[(n * DD + d) * II + i];   // [n][i][d]
    Wo_t[(n * DD + d) * II + i] = (__bf16)Wo[(n * II + i) * DD + d];   // [n][d][i]
  }
  if (t < DD * II) {
    int d = t / II, i = t - d * II;
    Wk_t[i * DD + d] = (__bf16)Wk[t];
    Wv_t[i * DD + d] = (__bf16)Wv[t];
  }
}

// ---------------- Kernel 2: Q projection (per-domain Wq) ----------------
// one wave: 16 rows of T x 64 cols (I), K-loop over D
__global__ void proj_q(const float* __restrict__ x, const __bf16* __restrict__ Wq_t,
                       __bf16* __restrict__ q_ws) {
  int wave = (blockIdx.x * blockDim.x + threadIdx.x) >> 5;
  int lane = threadIdx.x & 31;
  int tt = wave & 63;
  int nb = wave >> 6;                 // n*B + b
  if (nb >= NN * BB) return;
  int n = nb >> 2;
  int half = (lane >= 16) ? 1 : 0;
  int l16 = lane & 15;

  v8f acc[4];
  #pragma unroll
  for (int c = 0; c < 4; ++c) acc[c] = v8zero();

  const float* xrow = x + ((size_t)nb * TT + tt * 16 + l16) * DD + (half ? 8 : 0);
  const __bf16* wbase = Wq_t + ((size_t)n * II + l16) * DD + (half ? 8 : 0);
  for (int kb = 0; kb < DD; kb += 32) {
    v16bf bt[4];
    #pragma unroll
    for (int c = 0; c < 4; ++c) bt[c] = ldbf16x16(wbase + (size_t)c * 16 * DD + kb);
    v16bf a = a_from_f32(xrow + kb);
    #pragma unroll
    for (int c = 0; c < 4; ++c) acc[c] = wmma_bf16(a, bt[c], acc[c]);
  }
  #pragma unroll
  for (int c = 0; c < 4; ++c) {
    int i = c * 16 + l16, h = i >> 5, dh = i & 31;
    #pragma unroll
    for (int r = 0; r < 8; ++r) {
      int t = tt * 16 + r + 8 * half;
      q_ws[(((size_t)nb * HH + h) * TT + t) * DH + dh] = (__bf16)acc[c][r];
    }
  }
}

// ---------------- Kernel 3: K,V projection (shared Wk/Wv), fused ----------------
__global__ void proj_kv(const float* __restrict__ hist, const __bf16* __restrict__ Wk_t,
                        const __bf16* __restrict__ Wv_t,
                        __bf16* __restrict__ k_ws, __bf16* __restrict__ vt_ws) {
  int wave = (blockIdx.x * blockDim.x + threadIdx.x) >> 5;
  int lane = threadIdx.x & 31;
  const int STILES = SS / 16;         // 192
  int st = wave % STILES;
  int nb = wave / STILES;
  if (nb >= NN * BB) return;
  int n = nb >> 2, b = nb & 3;
  int half = (lane >= 16) ? 1 : 0;
  int l16 = lane & 15;

  int s = st * 16 + l16;              // this lane's A row
  int p = s >> 10, tsrc = s & (TT - 1);
  const float* arow = hist + ((((size_t)p * NN + n) * BB + b) * TT + tsrc) * DD + (half ? 8 : 0);
  const __bf16* wkb = Wk_t + (size_t)l16 * DD + (half ? 8 : 0);
  const __bf16* wvb = Wv_t + (size_t)l16 * DD + (half ? 8 : 0);

  v8f ak[4], av[4];
  #pragma unroll
  for (int c = 0; c < 4; ++c) { ak[c] = v8zero(); av[c] = v8zero(); }

  for (int kb = 0; kb < DD; kb += 32) {
    v16bf bk[4], bv[4];
    #pragma unroll
    for (int c = 0; c < 4; ++c) {
      bk[c] = ldbf16x16(wkb + (size_t)c * 16 * DD + kb);
      bv[c] = ldbf16x16(wvb + (size_t)c * 16 * DD + kb);
    }
    v16bf a = a_from_f32(arow + kb);
    #pragma unroll
    for (int c = 0; c < 4; ++c) {
      ak[c] = wmma_bf16(a, bk[c], ak[c]);
      av[c] = wmma_bf16(a, bv[c], av[c]);
    }
  }
  #pragma unroll
  for (int c = 0; c < 4; ++c) {
    int i = c * 16 + l16, h = i >> 5, dh = i & 31;
    #pragma unroll
    for (int r = 0; r < 8; ++r) {
      int srow = st * 16 + r + 8 * half;
      k_ws [(((size_t)nb * HH + h) * SS + srow) * DH + dh] = (__bf16)ak[c][r];   // [s][dh]
      vt_ws[(((size_t)nb * HH + h) * DH + dh) * SS + srow] = (__bf16)av[c][r];   // [dh][s]
    }
  }
}

// ---------------- Kernel 4: flash attention ----------------
// one wave per (n,b,h, 16-row t-tile); S processed in 128-col chunks
__global__ void attn(const __bf16* __restrict__ q_ws, const __bf16* __restrict__ k_ws,
                     const __bf16* __restrict__ vt_ws, const int* __restrict__ mask,
                     __bf16* __restrict__ o_ws) {
  __shared__ __align__(16) __bf16 Plds[4][16][128];   // per-wave P panel (bf16)
  int wid  = threadIdx.x >> 5;
  int lane = threadIdx.x & 31;
  int wave = blockIdx.x * 4 + wid;
  int tt  = wave & 63;
  int nbh = wave >> 6;                // n*B*H flattened: (n*B+b)*H + h
  if (nbh >= NN * BB * HH) return;
  int h  = nbh & 1;
  int nb = nbh >> 1;
  int b  = nb & 3;
  int half = (lane >= 16) ? 1 : 0;
  int l16 = lane & 15;

  const __bf16* kbase = k_ws  + (size_t)nbh * SS * DH;
  const __bf16* vbase = vt_ws + (size_t)nbh * DH * SS;

  // Q tile (A operand, resident for whole S loop)
  int tq = tt * 16 + l16;
  v16bf qa = a_from_bf16(q_ws + ((size_t)nbh * TT + tq) * DH + (half ? 8 : 0));

  float m[8], lsum[8];
  #pragma unroll
  for (int r = 0; r < 8; ++r) { m[r] = -1e30f; lsum[r] = 0.f; }
  v8f o0 = v8zero(), o1 = v8zero();
  const float scale = 0.17677669529663687f;   // 1/sqrt(32)
  const v8f cz = v8zero();

  for (int c = 0; c < SS / 128; ++c) {
    int sbase = c * 128;
    if (c + 1 < SS / 128)
      __builtin_prefetch(kbase + (size_t)(sbase + 128 + l16) * DH, 0, 0);

    // ---- scores: 8 tiles of 16 cols (batched loads, then WMMA group) ----
    v8f sc[8];
    float bias[8];
    #pragma unroll
    for (int g = 0; g < 2; ++g) {
      v16bf bk[4];
      #pragma unroll
      for (int j = 0; j < 4; ++j) {
        int jj = g * 4 + j;
        int scol = sbase + jj * 16 + l16;
        bk[j] = ldbf16x16(kbase + (size_t)scol * DH + (half ? 8 : 0));
        bias[jj] = mask[b * TT + (scol & (TT - 1))] ? 0.f : -10000.f;
      }
      #pragma unroll
      for (int j = 0; j < 4; ++j) sc[g * 4 + j] = wmma_bf16(qa, bk[j], cz);
    }

    // ---- online softmax per row (row r+8*half lives across the 16-lane group) ----
    float alpha[8];
    #pragma unroll
    for (int r = 0; r < 8; ++r) {
      float mx = -1e30f;
      #pragma unroll
      for (int j = 0; j < 8; ++j) {
        float v = sc[j][r] * scale + bias[j];
        sc[j][r] = v;
        mx = fmaxf(mx, v);
      }
      #pragma unroll
      for (int o = 1; o < 16; o <<= 1) mx = fmaxf(mx, __shfl_xor(mx, o, 32));
      float mn = fmaxf(m[r], mx);
      alpha[r] = __expf(m[r] - mn);
      m[r] = mn;
      float sum = 0.f;
      int trow = r + 8 * half;
      #pragma unroll
      for (int j = 0; j < 8; ++j) {
        float pe = __expf(sc[j][r] - mn);
        sum += pe;
        Plds[wid][trow][j * 16 + l16] = (__bf16)pe;     // C-layout -> LDS
      }
      #pragma unroll
      for (int o = 1; o < 16; o <<= 1) sum += __shfl_xor(sum, o, 32);
      lsum[r] = lsum[r] * alpha[r] + sum;
    }
    #pragma unroll
    for (int r = 0; r < 8; ++r) { o0[r] *= alpha[r]; o1[r] *= alpha[r]; }

    // all lanes' LDS writes must land before transposed re-read (same wave)
    asm volatile("s_wait_dscnt 0" ::: "memory");

    // ---- P @ V : reload P in A-layout, V^T columns contiguous ----
    #pragma unroll
    for (int kk = 0; kk < 4; ++kk) {
      const __bf16* pp = &Plds[wid][l16][kk * 32 + (half ? 8 : 0)];
      v16bf pa = a_from_bf16(pp);
      int sv = sbase + kk * 32 + (half ? 8 : 0);
      v16bf b0 = ldbf16x16(vbase + (size_t)l16 * SS + sv);
      v16bf b1 = ldbf16x16(vbase + (size_t)(l16 + 16) * SS + sv);
      o0 = wmma_bf16(pa, b0, o0);
      o1 = wmma_bf16(pa, b1, o1);
    }
  }

  // ---- finalize: divide by row sums, store [n][b][t][i] bf16 ----
  #pragma unroll
  for (int r = 0; r < 8; ++r) {
    float inv = 1.f / lsum[r];
    int t = tt * 16 + r + 8 * half;
    int i0 = h * 32 + l16;
    size_t base = ((size_t)nb * TT + t) * II;
    o_ws[base + i0]      = (__bf16)(o0[r] * inv);
    o_ws[base + i0 + 16] = (__bf16)(o1[r] * inv);
  }
}

// ---------------- Kernel 5: out projection + residual ----------------
__global__ void out_proj(const __bf16* __restrict__ o_ws, const __bf16* __restrict__ Wo_t,
                         const float* __restrict__ x, float* __restrict__ y) {
  int wave = (blockIdx.x * blockDim.x + threadIdx.x) >> 5;
  int lane = threadIdx.x & 31;
  int cg  = wave % 12;                 // 12 col-groups of 64 over D=768
  int rem = wave / 12;
  int tt = rem & 63;
  int nb = rem >> 6;
  if (nb >= NN * BB) return;
  int n = nb >> 2;
  int half = (lane >= 16) ? 1 : 0;
  int l16 = lane & 15;

  v8f acc[4];
  #pragma unroll
  for (int c = 0; c < 4; ++c) acc[c] = v8zero();

  int t = tt * 16 + l16;
  const __bf16* arow = o_ws + ((size_t)nb * TT + t) * II + (half ? 8 : 0);
  const __bf16* wbase = Wo_t + ((size_t)n * DD + cg * 64 + l16) * II + (half ? 8 : 0);
  #pragma unroll
  for (int kb = 0; kb < II; kb += 32) {
    v16bf bt[4];
    #pragma unroll
    for (int c = 0; c < 4; ++c) bt[c] = ldbf16x16(wbase + (size_t)c * 16 * II + kb);
    v16bf a = a_from_bf16(arow + kb);
    #pragma unroll
    for (int c = 0; c < 4; ++c) acc[c] = wmma_bf16(a, bt[c], acc[c]);
  }
  #pragma unroll
  for (int c = 0; c < 4; ++c) {
    int d = cg * 64 + c * 16 + l16;
    #pragma unroll
    for (int r = 0; r < 8; ++r) {
      int trow = tt * 16 + r + 8 * half;
      size_t idx = ((size_t)nb * TT + trow) * DD + d;
      y[idx] = acc[c][r] + x[idx];     // residual
    }
  }
}

// ---------------- Kernel 6: LayerNorm (in-place on y) ----------------
__global__ void layernorm(float* __restrict__ y, const float* __restrict__ gamma,
                          const float* __restrict__ beta) {
  int wave = (blockIdx.x * blockDim.x + threadIdx.x) >> 5;
  int lane = threadIdx.x & 31;
  if (wave >= NN * BB * TT) return;
  int n = wave / (BB * TT);
  float* row = y + (size_t)wave * DD;
  float s = 0.f, s2 = 0.f;
  float v[24];
  #pragma unroll
  for (int k = 0; k < 24; ++k) {
    float f = row[lane + 32 * k];
    v[k] = f; s += f; s2 += f * f;
  }
  #pragma unroll
  for (int o = 1; o < 32; o <<= 1) { s += __shfl_xor(s, o, 32); s2 += __shfl_xor(s2, o, 32); }
  float mu = s * (1.f / DD);
  float var = s2 * (1.f / DD) - mu * mu;
  float rstd = rsqrtf(var + LN_EPS);
  #pragma unroll
  for (int k = 0; k < 24; ++k) {
    int d = lane + 32 * k;
    row[d] = (v[k] - mu) * rstd * gamma[n * DD + d] + beta[n * DD + d];
  }
}

extern "C" void kernel_launch(void* const* d_in, const int* in_sizes, int n_in,
                              void* d_out, int out_size, void* d_ws, size_t ws_size,
                              hipStream_t stream) {
  const float* x     = (const float*)d_in[0];
  const float* hist  = (const float*)d_in[1];
  const int*   maskp = (const int*)  d_in[2];
  const float* Wq    = (const float*)d_in[3];
  const float* Wk    = (const float*)d_in[4];
  const float* Wv    = (const float*)d_in[5];
  const float* Wo    = (const float*)d_in[6];
  const float* gamma = (const float*)d_in[7];
  const float* beta  = (const float*)d_in[8];
  float* out = (float*)d_out;
  (void)in_sizes; (void)n_in; (void)out_size; (void)ws_size;

  unsigned char* w = (unsigned char*)d_ws;
  auto alloc = [&](size_t bytes) -> __bf16* {
    __bf16* p = (__bf16*)w;
    w += (bytes + 255) & ~(size_t)255;
    return p;
  };
  __bf16* Wq_t  = alloc((size_t)NN * II * DD * 2);
  __bf16* Wk_t  = alloc((size_t)II * DD * 2);
  __bf16* Wv_t  = alloc((size_t)II * DD * 2);
  __bf16* Wo_t  = alloc((size_t)NN * DD * II * 2);
  __bf16* q_ws  = alloc((size_t)NN * BB * HH * TT * DH * 2);
  __bf16* k_ws  = alloc((size_t)NN * BB * HH * SS * DH * 2);
  __bf16* vt_ws = alloc((size_t)NN * BB * HH * DH * SS * 2);
  __bf16* o_ws  = alloc((size_t)NN * BB * TT * II * 2);

  prep_weights<<<(NN * DD * II + 255) / 256, 256, 0, stream>>>(
      Wq, Wk, Wv, Wo, Wq_t, Wk_t, Wv_t, Wo_t);
  proj_q<<<(NN * BB * 64) / 4, 128, 0, stream>>>(x, Wq_t, q_ws);
  proj_kv<<<(NN * BB * (SS / 16)) / 4, 128, 0, stream>>>(hist, Wk_t, Wv_t, k_ws, vt_ws);
  attn<<<(NN * BB * HH * 64) / 4, 128, 0, stream>>>(q_ws, k_ws, vt_ws, maskp, o_ws);
  out_proj<<<(NN * BB * 64 * 12) / 4, 128, 0, stream>>>(o_ws, Wo_t, x, out);
  layernorm<<<(NN * BB * TT) / 4, 128, 0, stream>>>(out, gamma, beta);
}
